// MyGCN_4217657884863
// MI455X (gfx1250) — matrix operand order repaired and checked by reading, not compiled, not run on hardware
//
#include <hip/hip_runtime.h>

typedef __attribute__((ext_vector_type(16))) _Float16 v16h;
typedef __attribute__((ext_vector_type(8)))  float    v8f;

#define N_NODES    50000
#define N_EDGES    800000
#define DIM        64
#define NUM_GRAPHS 64

union AFrag { v16h v; _Float16 h[16]; };
union CFrag { v8f  v; float    f[8];  };

// ---------------------------------------------------------------------------
// degree / normalization
// ---------------------------------------------------------------------------
__global__ void zero_kernel(float* __restrict__ deg, float* __restrict__ pooled,
                            float* __restrict__ counts) {
    int i = blockIdx.x * blockDim.x + threadIdx.x;
    if (i < N_NODES) deg[i] = 0.0f;
    if (i < NUM_GRAPHS * DIM) pooled[i] = 0.0f;
    if (i < NUM_GRAPHS) counts[i] = 0.0f;
}

__global__ void deg_kernel(const int* __restrict__ dst, float* __restrict__ deg) {
    int e = blockIdx.x * blockDim.x + threadIdx.x;
    if (e < N_EDGES) unsafeAtomicAdd(&deg[dst[e]], 1.0f);
}

__global__ void rsqrt_kernel(float* __restrict__ deg_to_dis) {
    int i = blockIdx.x * blockDim.x + threadIdx.x;
    if (i < N_NODES) deg_to_dis[i] = rsqrtf(deg_to_dis[i] + 1.0f);
}

// ---------------------------------------------------------------------------
// Fused WMMA GEMM + self-loop/bias epilogue:
//   h   = (doRelu ? relu(X) : X) @ W
//   acc = h * dis^2 + b          (ready for the edge scatter to add into)
// 8 waves / block, each wave owns a 16-row stripe, 4 N-tiles, K = 2 x 32.
// A loads are unconditional from a clamped row pointer: an invalid lane's
// A-row only feeds output rows that are themselves invalid (store-guarded).
// Epilogue has a wave-uniform fast path (full 16-row tile) with no per-element
// bounds compares; only the final partial tile takes the guarded path.
// ---------------------------------------------------------------------------
__global__ __launch_bounds__(256) void gemm_fused_kernel(
        const float* __restrict__ X, const float* __restrict__ W,
        const float* __restrict__ bias, const float* __restrict__ dis,
        float* __restrict__ H, float* __restrict__ ACC, int doRelu) {
    __shared__ _Float16 Wt[DIM * DIM];   // Wt[n*64 + k] = (f16) W[k*64 + n]

    int tid = threadIdx.x;
    for (int idx = tid; idx < DIM * DIM; idx += 256) {
        int k = idx >> 6, n = idx & 63;
        Wt[n * DIM + k] = (_Float16)W[idx];    // coalesced global read
    }
    __syncthreads();

    int wave   = tid >> 5;
    int lane   = tid & 31;
    int lane16 = lane & 15;
    int hi     = lane >> 4;                    // lane half selector
    int rowBase = blockIdx.x * 128 + wave * 16;

    int  M = rowBase + lane16;
    const float* xrow = X + (size_t)(M < N_NODES ? M : 0) * DIM;

    // ---- hoist all A data: per kstep, contiguous runs [kb, kb+8) and
    //      [kb+16, kb+24) => 2x float4 each; 8 b128 loads total ----
    float4 p[2][4];
    #pragma unroll
    for (int ks = 0; ks < 2; ++ks) {
        int kb = ks * 32 + hi * 8;
        p[ks][0] = *(const float4*)(xrow + kb + 0);
        p[ks][1] = *(const float4*)(xrow + kb + 4);
        p[ks][2] = *(const float4*)(xrow + kb + 16);
        p[ks][3] = *(const float4*)(xrow + kb + 20);
    }
    if (doRelu) {   // fused input ReLU (previous layer's activation)
        #pragma unroll
        for (int ks = 0; ks < 2; ++ks)
            #pragma unroll
            for (int j = 0; j < 4; ++j) {
                p[ks][j].x = fmaxf(p[ks][j].x, 0.0f);
                p[ks][j].y = fmaxf(p[ks][j].y, 0.0f);
                p[ks][j].z = fmaxf(p[ks][j].z, 0.0f);
                p[ks][j].w = fmaxf(p[ks][j].w, 0.0f);
            }
    }

    CFrag acc[4];
    #pragma unroll
    for (int t = 0; t < 4; ++t)
        #pragma unroll
        for (int r = 0; r < 8; ++r) acc[t].f[r] = 0.0f;

    #pragma unroll
    for (int ks = 0; ks < 2; ++ks) {
        // ---- A fragment: 16x32 f16, ISA layout ----
        AFrag a;
        #pragma unroll
        for (int j = 0; j < 2; ++j) {
            a.h[4 * j + 0] = (_Float16)p[ks][j].x;
            a.h[4 * j + 1] = (_Float16)p[ks][j].y;
            a.h[4 * j + 2] = (_Float16)p[ks][j].z;
            a.h[4 * j + 3] = (_Float16)p[ks][j].w;
            a.h[8 + 4 * j + 0] = (_Float16)p[ks][2 + j].x;
            a.h[8 + 4 * j + 1] = (_Float16)p[ks][2 + j].y;
            a.h[8 + 4 * j + 2] = (_Float16)p[ks][2 + j].z;
            a.h[8 + 4 * j + 3] = (_Float16)p[ks][2 + j].w;
        }

        // ---- B fragments from LDS (transposed => contiguous K pairs) ----
        int kbaseB = ks * 32 + hi * 8;         // lanes 0-15: K 0..7,16..23; 16-31: K 8..15,24..31
        #pragma unroll
        for (int t = 0; t < 4; ++t) {
            AFrag b;
            const _Float16* wrow = Wt + (size_t)(t * 16 + lane16) * DIM + kbaseB;
            #pragma unroll
            for (int j = 0; j < 16; ++j) b.h[j] = wrow[j < 8 ? j : j + 8];
            acc[t].v = __builtin_amdgcn_wmma_f32_16x16x32_f16(
                false, a.v, false, b.v, (short)0, acc[t].v, false, false);
        }
    }

    // ---- fused epilogue.  D VGPR r -> row rowBase + hi*8 + r,
    //      col = t*16 + lane16.  Wave-uniform full-tile fast path. ----
    int rowTop = rowBase + hi * 8;
    if (rowBase + 16 <= N_NODES) {
        // fast path: no bounds checks, single base + constant offsets
        float dis2[8];
        #pragma unroll
        for (int r = 0; r < 8; ++r) {
            float d = dis[rowTop + r];
            dis2[r] = d * d;
        }
        #pragma unroll
        for (int t = 0; t < 4; ++t) {
            int col = t * 16 + lane16;
            float bc = bias[col];
            size_t base = (size_t)rowTop * DIM + col;
            #pragma unroll
            for (int r = 0; r < 8; ++r) {
                float hv = acc[t].f[r];
                H[base + (size_t)r * DIM]   = hv;
                ACC[base + (size_t)r * DIM] = hv * dis2[r] + bc;
            }
        }
    } else {
        float dis2[8];
        #pragma unroll
        for (int r = 0; r < 8; ++r) {
            int row = rowTop + r;
            float d = dis[row < N_NODES ? row : 0];
            dis2[r] = d * d;
        }
        #pragma unroll
        for (int t = 0; t < 4; ++t) {
            int col = t * 16 + lane16;
            float bc = bias[col];
            #pragma unroll
            for (int r = 0; r < 8; ++r) {
                int row = rowTop + r;
                if (row < N_NODES) {
                    float hv = acc[t].f[r];
                    size_t o = (size_t)row * DIM + col;
                    H[o]   = hv;
                    ACC[o] = hv * dis2[r] + bc;
                }
            }
        }
    }
}

// ---------------------------------------------------------------------------
// edge scatter: 16 lanes per edge, float4 each (covers 64 cols, 256B/row)
// ---------------------------------------------------------------------------
__global__ void scatter_kernel(const float4* __restrict__ h4,
                               const float* __restrict__ dis,
                               const int* __restrict__ src,
                               const int* __restrict__ dst,
                               float* __restrict__ acc) {
    int idx = blockIdx.x * blockDim.x + threadIdx.x;
    int e   = idx >> 4;
    int sub = idx & 15;
    if (e < N_EDGES) {
        int s = src[e], d = dst[e];
        float nrm = dis[s] * dis[d];
        float4 v = h4[s * 16 + sub];
        float* o = acc + (size_t)d * DIM + sub * 4;
        unsafeAtomicAdd(o + 0, v.x * nrm);
        unsafeAtomicAdd(o + 1, v.y * nrm);
        unsafeAtomicAdd(o + 2, v.z * nrm);
        unsafeAtomicAdd(o + 3, v.w * nrm);
    }
}

// ---------------------------------------------------------------------------
// global mean pool (sums + counts via L2 atomics); final ReLU fused here
// ---------------------------------------------------------------------------
__global__ void pool_kernel(const float* __restrict__ h,
                            const int* __restrict__ batch,
                            float* __restrict__ pooled,
                            float* __restrict__ counts) {
    int i = blockIdx.x * blockDim.x + threadIdx.x;
    if (i < N_NODES * DIM) {
        int node = i >> 6, c = i & 63;
        int g = batch[node];
        unsafeAtomicAdd(&pooled[g * DIM + c], fmaxf(h[i], 0.0f));
        if (c == 0) unsafeAtomicAdd(&counts[g], 1.0f);
    }
}

__global__ void final_kernel(const float* __restrict__ pooled,
                             const float* __restrict__ counts,
                             const float* __restrict__ Wl,
                             const float* __restrict__ bl,
                             float* __restrict__ out) {
    int g = threadIdx.x;
    if (g < NUM_GRAPHS) {
        float inv = 1.0f / fmaxf(counts[g], 1.0f);
        float s = 0.0f;
        #pragma unroll 8
        for (int c = 0; c < DIM; ++c) s += pooled[g * DIM + c] * inv * Wl[c];
        out[g] = s + bl[0];
    }
}

// ---------------------------------------------------------------------------
extern "C" void kernel_launch(void* const* d_in, const int* in_sizes, int n_in,
                              void* d_out, int out_size, void* d_ws, size_t ws_size,
                              hipStream_t stream) {
    const float* x     = (const float*)d_in[0];
    const int*   edge  = (const int*)d_in[1];    // [2, E]
    const int*   batch = (const int*)d_in[2];
    const float* W1 = (const float*)d_in[3];
    const float* b1 = (const float*)d_in[4];
    const float* W2 = (const float*)d_in[5];
    const float* b2 = (const float*)d_in[6];
    const float* W3 = (const float*)d_in[7];
    const float* b3 = (const float*)d_in[8];
    const float* Wl = (const float*)d_in[9];
    const float* bl = (const float*)d_in[10];
    const int* src = edge;
    const int* dst = edge + N_EDGES;

    // workspace (floats): dis[N] | h[N*64] | acc0[N*64] | acc1[N*64] |
    //                     pooled[G*64] | counts[G]
    float* ws     = (float*)d_ws;
    float* dis    = ws;
    float* h      = dis + N_NODES;
    float* acc0   = h + (size_t)N_NODES * DIM;
    float* acc1   = acc0 + (size_t)N_NODES * DIM;
    float* pooled = acc1 + (size_t)N_NODES * DIM;
    float* counts = pooled + NUM_GRAPHS * DIM;

    const int NE = N_NODES * DIM;

    // degrees -> deg_inv_sqrt (in place in `dis`)
    zero_kernel<<<(N_NODES + 255) / 256, 256, 0, stream>>>(dis, pooled, counts);
    deg_kernel<<<(N_EDGES + 255) / 256, 256, 0, stream>>>(dst, dis);
    rsqrt_kernel<<<(N_NODES + 255) / 256, 256, 0, stream>>>(dis);

    dim3 gemmGrid((N_NODES + 127) / 128);
    const float* Ws[3] = {W1, W2, W3};
    const float* bs[3] = {b1, b2, b3};
    float* accbuf[2] = {acc0, acc1};

    for (int l = 0; l < 3; ++l) {
        const float* in = (l == 0) ? x : accbuf[(l + 1) & 1];  // prev layer (pre-relu)
        float* out = accbuf[l & 1];
        gemm_fused_kernel<<<gemmGrid, 256, 0, stream>>>(
            in, Ws[l], bs[l], dis, h, out, l > 0);
        scatter_kernel<<<(N_EDGES * 16 + 255) / 256, 256, 0, stream>>>(
            (const float4*)h, dis, src, dst, out);
    }

    // layer 3 result lives in acc0 (l=2 -> l&1 == 0); relu fused into pooling
    pool_kernel<<<(NE + 255) / 256, 256, 0, stream>>>(acc0, batch, pooled, counts);
    final_kernel<<<1, 64, 0, stream>>>(pooled, counts, Wl, bl, (float*)d_out);
}
